// FakeFusedMoE_56014963474859
// MI455X (gfx1250) — matrix operand order repaired and checked by reading, not compiled, not run on hardware
//
#include <hip/hip_runtime.h>

typedef unsigned int u32;
typedef unsigned short u16;

typedef __bf16 bf16;
typedef __attribute__((ext_vector_type(16))) __bf16 bf16x16;
typedef __attribute__((ext_vector_type(8)))  __bf16 bf16x8;
typedef __attribute__((ext_vector_type(8)))  float  f32x8;

constexpr int E_ = 8;
constexpr int H_ = 1024;
constexpr int F_ = 1024;
constexpr int NTOK = 2048;   // B*T
constexpr int LDSPAD = 40;   // 32 bf16 + 8 pad -> 80B row stride: conflict-free, 16B aligned

static __device__ __forceinline__ u16 f32_to_bf16(float f) {
  u32 u = __float_as_uint(f);
  u += 0x7fffu + ((u >> 16) & 1u);   // round-to-nearest-even
  return (u16)(u >> 16);
}

// --------- CDNA5 async global->LDS copy (ASYNCcnt-tracked DMA path) ---------
// VDST = per-lane LDS byte offset (low 32 bits of generic LDS address; the
// aperture lives in the high half), VADDR = 64-bit global address.
static __device__ __forceinline__ void async_copy_b128(const u16* gsrc, bf16* ldst) {
  u32 l = (u32)(uintptr_t)ldst;
  unsigned long long g = (unsigned long long)(uintptr_t)gsrc;
  asm volatile("global_load_async_to_lds_b128 %0, %1, off"
               :: "v"(l), "v"(g) : "memory");
}
static __device__ __forceinline__ void wait_async() {
  asm volatile("s_wait_asynccnt 0x0" ::: "memory");
}

// ---------------- fp32 -> bf16 bulk convert (4 elems / thread) ----------------
__global__ void moe_cvt_bf16_kernel(const float* __restrict__ src,
                                    u16* __restrict__ dst, int n4) {
  int i = blockIdx.x * blockDim.x + threadIdx.x;
  if (i >= n4) return;
  float4 v = ((const float4*)src)[i];
  ushort4 o;
  o.x = f32_to_bf16(v.x); o.y = f32_to_bf16(v.y);
  o.z = f32_to_bf16(v.z); o.w = f32_to_bf16(v.w);
  ((ushort4*)dst)[i] = o;
}

// ---------------- router: logits -> softmax -> top2 -> renorm ----------------
__global__ __launch_bounds__(256) void moe_router_kernel(
    const float* __restrict__ h,        // [NTOK][H]
    const float* __restrict__ rw,       // [E][H]
    float* __restrict__ combine) {      // [E][NTOK]
  const int lane = threadIdx.x & 31;
  const int wave = threadIdx.x >> 5;
  const int tok = blockIdx.x * 8 + wave;

  float acc[E_];
#pragma unroll
  for (int e = 0; e < E_; ++e) acc[e] = 0.f;

  const float* hrow = h + (size_t)tok * H_;
  for (int k = lane; k < H_; k += 32) {
    float x = hrow[k];
#pragma unroll
    for (int e = 0; e < E_; ++e) acc[e] += x * rw[e * H_ + k];
  }
#pragma unroll
  for (int e = 0; e < E_; ++e) {
#pragma unroll
    for (int off = 16; off > 0; off >>= 1)
      acc[e] += __shfl_xor(acc[e], off, 32);
  }
  float mx = acc[0];
#pragma unroll
  for (int e = 1; e < E_; ++e) mx = fmaxf(mx, acc[e]);
  float p[E_];
#pragma unroll
  for (int e = 0; e < E_; ++e) p[e] = __expf(acc[e] - mx);
  int i1 = 0;
#pragma unroll
  for (int e = 1; e < E_; ++e) if (p[e] > p[i1]) i1 = e;
  int i2 = (i1 == 0) ? 1 : 0;
#pragma unroll
  for (int e = 0; e < E_; ++e) if (e != i1 && p[e] > p[i2]) i2 = e;
  float denom = p[i1] + p[i2];
  float w1 = p[i1] / denom, w2 = p[i2] / denom;
  if (lane < E_) {
    float w = (lane == i1) ? w1 : ((lane == i2) ? w2 : 0.f);
    combine[lane * NTOK + tok] = w;
  }
}

// ---------------- WMMA fragment loaders (ISA 7.12.2 layouts) ----------------
static __device__ __forceinline__ bf16x16 load_frag_a(const bf16* row, int khalf) {
  bf16x8 lo = *(const bf16x8*)(row + khalf * 8);
  bf16x8 hi = *(const bf16x8*)(row + 16 + khalf * 8);
  return __builtin_shufflevector(lo, hi, 0,1,2,3,4,5,6,7,8,9,10,11,12,13,14,15);
}
static __device__ __forceinline__ bf16x16 load_frag_b(const bf16* row, int khalf) {
  bf16x8 lo = *(const bf16x8*)(row + khalf * 8);
  bf16x8 hi = *(const bf16x8*)(row + 16 + khalf * 8);
  return __builtin_shufflevector(lo, hi, 0,1,2,3,4,5,6,7,8,9,10,11,12,13,14,15);
}

// ---------------- GEMM1: inter = combine * silu(h@Wg^T) * (h@Wu^T) ----------------
// Block tile: 64 tokens x 128 gate cols (+128 paired up cols).
// 8 waves = 2(m) x 4(n); each wave owns a 32x32 gate region AND the matching
// 32x32 up region: 8 WMMAs per K-step from 2 A-frags + 2 Bg-frags + 2 Bu-frags.
// Tiles arrive via async global->LDS DMA (ASYNCcnt), not through VGPRs.
__global__ __launch_bounds__(256) void moe_gemm1_kernel(
    const u16* __restrict__ hb,      // [NTOK][H] bf16
    const u16* __restrict__ gub,     // [2F][H] bf16 (this expert)
    const float* __restrict__ comb,  // [NTOK] (this expert)
    u16* __restrict__ inter) {       // [NTOK][F] bf16
  __shared__ __align__(16) bf16 As[64 * LDSPAD];
  __shared__ __align__(16) bf16 Bg[128 * LDSPAD];
  __shared__ __align__(16) bf16 Bu[128 * LDSPAD];

  const int tid = threadIdx.x;
  const int lane = tid & 31, wave = tid >> 5;
  const int ln = lane & 15, khalf = lane >> 4;
  const int mi = wave >> 2, ni = wave & 3;
  const int m0 = blockIdx.y * 64;
  const int c0 = blockIdx.x * 128;

  f32x8 accG[2][2] = {{{}, {}}, {{}, {}}};
  f32x8 accU[2][2] = {{{}, {}}, {{}, {}}};

  const int arow = tid >> 2, aco = (tid & 3) * 8;   // A: 64x32, 8 elems/thread
  const int brow = tid >> 1, bco = (tid & 1) * 16;  // B: 128x32, 16 elems/thread

  for (int k0 = 0; k0 < H_; k0 += 32) {
    __syncthreads();   // previous K-step's consumers are done with the tiles
    async_copy_b128(hb + (size_t)(m0 + arow) * H_ + k0 + aco,
                    As + arow * LDSPAD + aco);
    async_copy_b128(gub + (size_t)(c0 + brow) * H_ + k0 + bco,
                    Bg + brow * LDSPAD + bco);
    async_copy_b128(gub + (size_t)(c0 + brow) * H_ + k0 + bco + 8,
                    Bg + brow * LDSPAD + bco + 8);
    async_copy_b128(gub + (size_t)(F_ + c0 + brow) * H_ + k0 + bco,
                    Bu + brow * LDSPAD + bco);
    async_copy_b128(gub + (size_t)(F_ + c0 + brow) * H_ + k0 + bco + 8,
                    Bu + brow * LDSPAD + bco + 8);
    if (k0 + 32 < H_) {  // global_prefetch_b8 of next B tiles (L2-resident)
      __builtin_prefetch(gub + (size_t)(c0 + brow) * H_ + k0 + 32 + bco, 0, 1);
      __builtin_prefetch(gub + (size_t)(F_ + c0 + brow) * H_ + k0 + 32 + bco, 0, 1);
    }
    wait_async();      // my DMA writes have landed in LDS
    __syncthreads();   // everyone's tiles are published

    bf16x16 a[2], bg[2], bu[2];
#pragma unroll
    for (int i = 0; i < 2; ++i)
      a[i] = load_frag_a(As + (mi * 32 + i * 16 + ln) * LDSPAD, khalf);
#pragma unroll
    for (int j = 0; j < 2; ++j) {
      bg[j] = load_frag_b(Bg + (ni * 32 + j * 16 + ln) * LDSPAD, khalf);
      bu[j] = load_frag_b(Bu + (ni * 32 + j * 16 + ln) * LDSPAD, khalf);
    }
#pragma unroll
    for (int i = 0; i < 2; ++i)
#pragma unroll
      for (int j = 0; j < 2; ++j) {
        accG[i][j] = __builtin_amdgcn_wmma_f32_16x16x32_bf16(
            false, a[i], false, bg[j], (short)0, accG[i][j], false, false);
        accU[i][j] = __builtin_amdgcn_wmma_f32_16x16x32_bf16(
            false, a[i], false, bu[j], (short)0, accU[i][j], false, false);
      }
  }

  // epilogue: SwiGLU * combine weight, store bf16 intermediate
#pragma unroll
  for (int i = 0; i < 2; ++i) {
#pragma unroll
    for (int r = 0; r < 8; ++r) {
      const int m = m0 + mi * 32 + i * 16 + r + 8 * khalf;  // C VGPR r: rows r / r+8
      const float w = comb[m];
#pragma unroll
      for (int j = 0; j < 2; ++j) {
        const int col = c0 + ni * 32 + j * 16 + ln;
        const float g = accG[i][j][r];
        const float u = accU[i][j][r];
        const float silu = g / (1.f + __expf(-g));
        inter[(size_t)m * F_ + col] = f32_to_bf16(silu * u * w);
      }
    }
  }
}

// ---------------- GEMM2: out += inter @ down^T ----------------
// Block tile: 64 tokens x 128 hidden cols. 8 waves = 2(m) x 4(n); each wave
// owns a 32x32 C region: 4 WMMAs per K-step from 2 A-frags + 2 B-frags.
__global__ __launch_bounds__(256) void moe_gemm2_kernel(
    const u16* __restrict__ inter,   // [NTOK][F] bf16
    const u16* __restrict__ dpb,     // [H][F] bf16 (this expert)
    float* __restrict__ out) {       // [NTOK][H] fp32, accumulated
  __shared__ __align__(16) bf16 As[64 * LDSPAD];
  __shared__ __align__(16) bf16 Bs[128 * LDSPAD];

  const int tid = threadIdx.x;
  const int lane = tid & 31, wave = tid >> 5;
  const int ln = lane & 15, khalf = lane >> 4;
  const int mi = wave >> 2, ni = wave & 3;
  const int m0 = blockIdx.y * 64;
  const int c0 = blockIdx.x * 128;

  f32x8 acc[2][2] = {{{}, {}}, {{}, {}}};

  const int arow = tid >> 2, aco = (tid & 3) * 8;   // A: 64x32, 8 elems/thread
  const int brow = tid >> 1, bco = (tid & 1) * 16;  // B: 128x32, 16 elems/thread

  for (int k0 = 0; k0 < F_; k0 += 32) {
    __syncthreads();
    async_copy_b128(inter + (size_t)(m0 + arow) * F_ + k0 + aco,
                    As + arow * LDSPAD + aco);
    async_copy_b128(dpb + (size_t)(c0 + brow) * F_ + k0 + bco,
                    Bs + brow * LDSPAD + bco);
    async_copy_b128(dpb + (size_t)(c0 + brow) * F_ + k0 + bco + 8,
                    Bs + brow * LDSPAD + bco + 8);
    if (k0 + 32 < F_)
      __builtin_prefetch(dpb + (size_t)(c0 + brow) * F_ + k0 + 32 + bco, 0, 1);
    wait_async();
    __syncthreads();

    bf16x16 a[2], b[2];
#pragma unroll
    for (int i = 0; i < 2; ++i)
      a[i] = load_frag_a(As + (mi * 32 + i * 16 + ln) * LDSPAD, khalf);
#pragma unroll
    for (int j = 0; j < 2; ++j)
      b[j] = load_frag_b(Bs + (ni * 32 + j * 16 + ln) * LDSPAD, khalf);
#pragma unroll
    for (int i = 0; i < 2; ++i)
#pragma unroll
      for (int j = 0; j < 2; ++j)
        acc[i][j] = __builtin_amdgcn_wmma_f32_16x16x32_bf16(
            false, a[i], false, b[j], (short)0, acc[i][j], false, false);
  }

#pragma unroll
  for (int i = 0; i < 2; ++i)
#pragma unroll
    for (int r = 0; r < 8; ++r) {
      const int m = m0 + mi * 32 + i * 16 + r + 8 * khalf;
#pragma unroll
      for (int j = 0; j < 2; ++j) {
        float* o = out + (size_t)m * H_ + c0 + ni * 32 + j * 16 + ln;
        *o += acc[i][j][r];
      }
    }
}

extern "C" void kernel_launch(void* const* d_in, const int* in_sizes, int n_in,
                              void* d_out, int out_size, void* d_ws, size_t ws_size,
                              hipStream_t stream) {
  const float* h  = (const float*)d_in[0];   // [2,1024,1024]
  const float* rw = (const float*)d_in[1];   // [8,1024]
  const float* gu = (const float*)d_in[2];   // [8,2048,1024]
  const float* dp = (const float*)d_in[3];   // [8,1024,1024]
  float* out = (float*)d_out;                // [2,1024,1024] fp32
  char* ws = (char*)d_ws;

  size_t off = 0;
  float* combine = (float*)(ws + off); off += (size_t)E_ * NTOK * sizeof(float);
  u16* hb    = (u16*)(ws + off); off += (size_t)NTOK * H_ * sizeof(u16);
  u16* gub   = (u16*)(ws + off); off += (size_t)E_ * 2 * F_ * H_ * sizeof(u16);
  u16* dpb   = (u16*)(ws + off); off += (size_t)E_ * H_ * F_ * sizeof(u16);
  u16* inter = (u16*)(ws + off); off += (size_t)NTOK * F_ * sizeof(u16);

  // one-time (per call) fp32 -> bf16 conversion; everything then lives in L2
  {
    int n4 = NTOK * H_ / 4;
    moe_cvt_bf16_kernel<<<(n4 + 255) / 256, 256, 0, stream>>>(h, hb, n4);
  }
  {
    int n4 = E_ * 2 * F_ * H_ / 4;
    moe_cvt_bf16_kernel<<<(n4 + 255) / 256, 256, 0, stream>>>(gu, gub, n4);
  }
  {
    int n4 = E_ * H_ * F_ / 4;
    moe_cvt_bf16_kernel<<<(n4 + 255) / 256, 256, 0, stream>>>(dp, dpb, n4);
  }

  moe_router_kernel<<<NTOK / 8, 256, 0, stream>>>(h, rw, combine);

  hipMemsetAsync(out, 0, (size_t)NTOK * H_ * sizeof(float), stream);

  for (int e = 0; e < E_; ++e) {
    moe_gemm1_kernel<<<dim3(F_ / 128, NTOK / 64), 256, 0, stream>>>(
        hb, gub + (size_t)e * 2 * F_ * H_, combine + (size_t)e * NTOK, inter);
    moe_gemm2_kernel<<<dim3(H_ / 128, NTOK / 64), 256, 0, stream>>>(
        inter, dpb + (size_t)e * H_ * F_, out);
  }
}
